// SingleDecoder_64158221467994
// MI455X (gfx1250) — matrix operand order repaired and checked by reading, not compiled, not run on hardware
//
#include <hip/hip_runtime.h>
#include <math.h>

// ---------------- model dims (fixed by the reference) ----------------
constexpr int S_ = 4, IN_ = 4096, H_ = 2048, D_ = 6, B_ = 1024, BOT_ = 128,
              IMG_ = 768, TXT_ = 768;

// ---------------- WMMA types ----------------
typedef __bf16 bf16;
typedef bf16  bf16x4  __attribute__((ext_vector_type(4)));
typedef bf16  bf16x8  __attribute__((ext_vector_type(8)));
typedef bf16  bf16x16 __attribute__((ext_vector_type(16)));
typedef float f32x8   __attribute__((ext_vector_type(8)));

#define BM 128
#define BN 128
#define BK 32
#define LDT 40   // LDS stride in bf16 elements (32 + 8 pad; 16B-aligned fragments)

__device__ __forceinline__ float gelu_exact(float x) {
  return 0.5f * x * (1.0f + erff(x * 0.70710678118654752440f));
}

// Workgroup barrier that drains ONLY the DS counter (LDS visibility) and
// leaves global loads (LOADcnt) in flight across the barrier -- unlike
// __syncthreads(), which the compiler lowers with s_wait_loadcnt_dscnt 0.
// Per ISA: each wave's dscnt==0 covers both its LDS spills (visible to the
// WGP) and its own fragment ds_loads (data already in VGPRs), so the idle
// buffer may be overwritten after the barrier.
__device__ __forceinline__ void wg_barrier_lds() {
  asm volatile(
      "s_wait_dscnt 0x0\n\t"
      "s_barrier_signal -1\n\t"
      "s_barrier_wait -1" ::: "memory");
}

enum { EPI_LINEAR = 0, EPI_GELU = 1, EPI_RES = 2 };

// ---------------------------------------------------------------------
// Deterministic per-subject routing: gidx[s*B + j] = j-th row of subject s
// ---------------------------------------------------------------------
__global__ __launch_bounds__(1024) void route_kernel(
    const int* __restrict__ sid, int* __restrict__ counts, int* __restrict__ gidx) {
  __shared__ int s_sid[B_];
  int t = threadIdx.x;
  s_sid[t] = sid[t];
  if (t < S_) counts[t] = 0;
  __syncthreads();
  int s = s_sid[t];
  int pos = 0;
  for (int j = 0; j < t; ++j) pos += (s_sid[j] == s) ? 1 : 0;
  gidx[s * B_ + pos] = t;
  atomicMax(&counts[s], pos + 1);
}

// ---------------------------------------------------------------------
// C[M,N] = epilogue( A[M,K] @ W[K,N] + bias )   (bf16 WMMA, fp32 accum)
// Double-buffered LDS + register prefetch: while tile t computes from
// LDS[buf], tile t+1 spills into LDS[buf^1] and tile t+2 streams from HBM.
// One DS-only barrier per K-tile; global prefetch stays in flight across it.
// ---------------------------------------------------------------------
template <int EPI>
__global__ __launch_bounds__(256) void gemm_bf16(
    const float* __restrict__ A, const float* __restrict__ W,
    const float* __restrict__ bias, const float* __restrict__ res,
    float* __restrict__ C, int M, const int* __restrict__ Mdyn, int N, int K,
    int lda, int ldc, int ldres, const int* __restrict__ gidx, int gatherA,
    int scatterC, int gatherRes) {
  const int Meff = Mdyn ? Mdyn[0] : M;
  const int tileM = (int)blockIdx.y * BM;
  if (tileM >= Meff) return;  // block-uniform: safe w.r.t. WMMA EXEC rule
  const int n0 = (int)blockIdx.x * BN;

  __shared__ alignas(16) bf16 Als[2][BM * LDT];  // row-major  [m][k]
  __shared__ alignas(16) bf16 Bls[2][BN * LDT];  // col-major  [n][k]

  const int tid = threadIdx.x;
  const int lane = tid & 31;
  const int wave = tid >> 5;
  const int waveM = wave & 3;  // 4 waves over M (32 rows each)
  const int waveN = wave >> 2; // 2 waves over N (64 cols each)
  const int lnid = lane & 15;
  const int half = (lane >> 4) & 1;

  // ---- hoisted per-thread staging offsets (K-invariant) ----
  // A tile: 128 rows x 32 cols = 1024 float4 -> 4 per thread
  // B tile:  32 rows x 128 cols = 1024 float4 -> 4 per thread
  int aOff[4];  // element offset of A float4 source (add k)
  int bOff[4];  // element offset of W float4 source (advances by BK*N)
  int aRow[4], aC4[4], bKrow[4], bC4[4];
#pragma unroll
  for (int p = 0; p < 4; ++p) {
    int idx = p * 256 + tid;
    aRow[p] = idx >> 3;               // 8 float4 per 32-col A row
    aC4[p] = (idx & 7) << 2;
    int mg = tileM + aRow[p];
    int ml = (mg < Meff) ? mg : (Meff - 1);
    int ra = gatherA ? gidx[ml] : ml;
    aOff[p] = ra * lda + aC4[p];

    bKrow[p] = idx >> 5;              // 32 float4 per 128-col B row
    bC4[p] = (idx & 31) << 2;
    bOff[p] = bKrow[p] * N + n0 + bC4[p];
  }

  f32x8 acc[2][4];
#pragma unroll
  for (int i = 0; i < 2; ++i)
#pragma unroll
    for (int j = 0; j < 4; ++j)
#pragma unroll
      for (int r = 0; r < 8; ++r) acc[i][j][r] = 0.0f;

  const int nt = K / BK;  // number of K-tiles (K is a multiple of 32)
  float4 aReg[4], bReg[4];

  // ---- prologue: tile 0 -> regs -> LDS[0]; tile 1 -> regs ----
#pragma unroll
  for (int p = 0; p < 4; ++p) aReg[p] = *(const float4*)(A + (size_t)aOff[p]);
#pragma unroll
  for (int p = 0; p < 4; ++p) bReg[p] = *(const float4*)(W + (size_t)bOff[p]);
#pragma unroll
  for (int p = 0; p < 4; ++p) {
    bf16x4 v;
    v[0] = (bf16)aReg[p].x; v[1] = (bf16)aReg[p].y;
    v[2] = (bf16)aReg[p].z; v[3] = (bf16)aReg[p].w;
    *(bf16x4*)(&Als[0][0] + (size_t)aRow[p] * LDT + aC4[p]) = v;
    Bls[0][(size_t)(bC4[p] + 0) * LDT + bKrow[p]] = (bf16)bReg[p].x;
    Bls[0][(size_t)(bC4[p] + 1) * LDT + bKrow[p]] = (bf16)bReg[p].y;
    Bls[0][(size_t)(bC4[p] + 2) * LDT + bKrow[p]] = (bf16)bReg[p].z;
    Bls[0][(size_t)(bC4[p] + 3) * LDT + bKrow[p]] = (bf16)bReg[p].w;
  }
  if (nt > 1) {
#pragma unroll
    for (int p = 0; p < 4; ++p)
      aReg[p] = *(const float4*)(A + (size_t)(aOff[p] + BK));
#pragma unroll
    for (int p = 0; p < 4; ++p) {
      bOff[p] += BK * N;
      bReg[p] = *(const float4*)(W + (size_t)bOff[p]);
    }
  }
  wg_barrier_lds();

  for (int t = 0; t < nt; ++t) {
    const int buf = t & 1;
    // ---- spill tile t+1 (in regs) into the idle LDS buffer ----
    if (t + 1 < nt) {
      bf16* Ad = &Als[buf ^ 1][0];
      bf16* Bd = &Bls[buf ^ 1][0];
#pragma unroll
      for (int p = 0; p < 4; ++p) {
        bf16x4 v;
        v[0] = (bf16)aReg[p].x; v[1] = (bf16)aReg[p].y;
        v[2] = (bf16)aReg[p].z; v[3] = (bf16)aReg[p].w;
        *(bf16x4*)(Ad + (size_t)aRow[p] * LDT + aC4[p]) = v;
        Bd[(size_t)(bC4[p] + 0) * LDT + bKrow[p]] = (bf16)bReg[p].x;
        Bd[(size_t)(bC4[p] + 1) * LDT + bKrow[p]] = (bf16)bReg[p].y;
        Bd[(size_t)(bC4[p] + 2) * LDT + bKrow[p]] = (bf16)bReg[p].z;
        Bd[(size_t)(bC4[p] + 3) * LDT + bKrow[p]] = (bf16)bReg[p].w;
      }
    }
    // ---- stream tile t+2 from HBM into regs (overlaps WMMA below) ----
    if (t + 2 < nt) {
      const int ka = (t + 2) * BK;
#pragma unroll
      for (int p = 0; p < 4; ++p)
        aReg[p] = *(const float4*)(A + (size_t)(aOff[p] + ka));
#pragma unroll
      for (int p = 0; p < 4; ++p) {
        bOff[p] += BK * N;
        bReg[p] = *(const float4*)(W + (size_t)bOff[p]);
      }
    }

    // ---- (2x4) WMMA fragments from LDS[buf] ----
    const bf16* As = &Als[buf][0];
    const bf16* Bs = &Bls[buf][0];
    bf16x16 aF[2], bF[4];
#pragma unroll
    for (int fm = 0; fm < 2; ++fm) {
      // A frag: lanes0-15 K=[0..7]&[16..23], lanes16-31 K=[8..15]&[24..31]
      const bf16* Ab = As + (size_t)(waveM * 32 + fm * 16 + lnid) * LDT +
                       half * 8;
      bf16x8 lo = *(const bf16x8*)Ab;
      bf16x8 hi = *(const bf16x8*)(Ab + 16);
      aF[fm] = __builtin_shufflevector(lo, hi, 0, 1, 2, 3, 4, 5, 6, 7, 8, 9,
                                       10, 11, 12, 13, 14, 15);
    }
#pragma unroll
    for (int fn = 0; fn < 4; ++fn) {
      // B frag: lanes0-15 K=[0..7]&[16..23], lanes16-31 K=[8..15]&[24..31]
      const bf16* Bb = Bs + (size_t)(waveN * 64 + fn * 16 + lnid) * LDT +
                       half * 8;
      bf16x8 lo = *(const bf16x8*)Bb;
      bf16x8 hi = *(const bf16x8*)(Bb + 16);
      bF[fn] = __builtin_shufflevector(lo, hi, 0, 1, 2, 3, 4, 5, 6, 7, 8, 9,
                                       10, 11, 12, 13, 14, 15);
    }
#pragma unroll
    for (int fm = 0; fm < 2; ++fm)
#pragma unroll
      for (int fn = 0; fn < 4; ++fn)
        acc[fm][fn] = __builtin_amdgcn_wmma_f32_16x16x32_bf16(
            false, aF[fm], false, bF[fn], (short)0, acc[fm][fn], false, false);

    if (t + 1 < nt) wg_barrier_lds();  // last tile: nothing reads LDS after
  }

  // ---- epilogue: bias (+gelu / +residual), masked scatter-store ----
  int cols[4];
  float bv[4];
#pragma unroll
  for (int fn = 0; fn < 4; ++fn) {
    cols[fn] = n0 + waveN * 64 + fn * 16 + lnid;
    bv[fn] = bias[cols[fn]];
  }
#pragma unroll
  for (int fm = 0; fm < 2; ++fm) {
#pragma unroll
    for (int r = 0; r < 8; ++r) {
      int mrow = tileM + waveM * 32 + fm * 16 + half * 8 + r;
      if (mrow < Meff) {
        int rc = scatterC ? gidx[mrow] : mrow;
        int rr = gatherRes ? gidx[mrow] : mrow;
#pragma unroll
        for (int fn = 0; fn < 4; ++fn) {
          float v = acc[fm][fn][r] + bv[fn];
          if (EPI == EPI_GELU) v = gelu_exact(v);
          if (EPI == EPI_RES) v += res[(size_t)rr * ldres + cols[fn]];
          C[(size_t)rc * ldc + cols[fn]] = v;
        }
      }
    }
  }
}

// ---------------------------------------------------------------------
// out[row] = (resid? resid[row] : 0) + gelu( ln(pre[row]; g,b) )
// ---------------------------------------------------------------------
__global__ __launch_bounds__(256) void ln_gelu_kernel(
    const float* __restrict__ pre, const float* __restrict__ gam,
    const float* __restrict__ bet, const int* __restrict__ sid,
    const float* __restrict__ resid, float* __restrict__ out, int n) {
  const int row = blockIdx.x;
  const float* pr = pre + (size_t)row * n;
  const int s = sid ? sid[row] : 0;
  const float* g = gam + (size_t)s * n;
  const float* b = bet + (size_t)s * n;

  __shared__ float red[256], red2[256];
  float sum = 0.0f, sq = 0.0f;
  for (int c = threadIdx.x; c < n; c += 256) {
    float v = pr[c];
    sum += v;
    sq += v * v;
  }
  red[threadIdx.x] = sum;
  red2[threadIdx.x] = sq;
  __syncthreads();
  for (int st = 128; st > 0; st >>= 1) {
    if ((int)threadIdx.x < st) {
      red[threadIdx.x] += red[threadIdx.x + st];
      red2[threadIdx.x] += red2[threadIdx.x + st];
    }
    __syncthreads();
  }
  float mean = red[0] / (float)n;
  float var = red2[0] / (float)n - mean * mean;
  var = var > 0.0f ? var : 0.0f;
  float inv = rsqrtf(var + 1e-5f);

  for (int c = threadIdx.x; c < n; c += 256) {
    float v = (pr[c] - mean) * inv * g[c] + b[c];
    v = gelu_exact(v);
    if (resid) v += resid[(size_t)row * n + c];
    out[(size_t)row * n + c] = v;
  }
}

// ---------------------------------------------------------------------
extern "C" void kernel_launch(void* const* d_in, const int* in_sizes, int n_in,
                              void* d_out, int out_size, void* d_ws,
                              size_t ws_size, hipStream_t stream) {
  const float* voxels    = (const float*)d_in[0];
  const int*   sid       = (const int*)d_in[1];
  const float* ad_down_w = (const float*)d_in[2];
  const float* ad_down_b = (const float*)d_in[3];
  const float* ad_up_w   = (const float*)d_in[4];
  const float* ad_up_b   = (const float*)d_in[5];
  const float* enc_w     = (const float*)d_in[6];
  const float* enc_b     = (const float*)d_in[7];
  const float* enc_ln_g  = (const float*)d_in[8];
  const float* enc_ln_b  = (const float*)d_in[9];
  const float* bb_w      = (const float*)d_in[10];
  const float* bb_b      = (const float*)d_in[11];
  const float* bb_ln_g   = (const float*)d_in[12];
  const float* bb_ln_b   = (const float*)d_in[13];
  const float* img_w     = (const float*)d_in[14];
  const float* img_b     = (const float*)d_in[15];
  const float* txt_w     = (const float*)d_in[16];
  const float* txt_b     = (const float*)d_in[17];
  float* out = (float*)d_out;

  char* ws = (char*)d_ws;
  int* counts = (int*)ws;                              // 4 ints
  int* gidx = (int*)(ws + 64);                         // S*B ints (16 KB)
  float* t_bot = (float*)(ws + 65536);                 // S*B*BOT  (2 MB)
  float* hbuf  = t_bot + (size_t)S_ * B_ * BOT_;       // B*IN     (16 MB)
  float* pre   = hbuf + (size_t)B_ * IN_;              // B*H      (8 MB)
  float* xbuf  = pre + (size_t)B_ * H_;                // B*H      (8 MB)

  route_kernel<<<1, 1024, 0, stream>>>(sid, counts, gidx);

  const int gy = B_ / BM;  // 8 row tiles (dynamic early-exit per subject)

  // 1) adapter down: t_bot[s] = gelu(voxels[g] @ dw_s + db_s)
  for (int s = 0; s < S_; ++s)
    gemm_bf16<EPI_GELU><<<dim3(BOT_ / BN, gy), 256, 0, stream>>>(
        voxels, ad_down_w + (size_t)s * IN_ * BOT_, ad_down_b + (size_t)s * BOT_,
        nullptr, t_bot + (size_t)s * B_ * BOT_, B_, counts + s, BOT_, IN_, IN_,
        BOT_, 0, gidx + s * B_, /*gatherA*/ 1, /*scatterC*/ 0, /*gatherRes*/ 0);

  // 2) adapter up + residual: h[g] = voxels[g] + t_bot[s] @ uw_s + ub_s
  for (int s = 0; s < S_; ++s)
    gemm_bf16<EPI_RES><<<dim3(IN_ / BN, gy), 256, 0, stream>>>(
        t_bot + (size_t)s * B_ * BOT_, ad_up_w + (size_t)s * BOT_ * IN_,
        ad_up_b + (size_t)s * IN_, voxels, hbuf, B_, counts + s, IN_, BOT_,
        BOT_, IN_, IN_, gidx + s * B_, /*gatherA*/ 0, /*scatterC*/ 1,
        /*gatherRes*/ 1);

  // 3) encoder linear: pre[g] = h[g] @ ew_s + eb_s
  for (int s = 0; s < S_; ++s)
    gemm_bf16<EPI_LINEAR><<<dim3(H_ / BN, gy), 256, 0, stream>>>(
        hbuf, enc_w + (size_t)s * IN_ * H_, enc_b + (size_t)s * H_, nullptr,
        pre, B_, counts + s, H_, IN_, IN_, H_, 0, gidx + s * B_, /*gatherA*/ 1,
        /*scatterC*/ 1, /*gatherRes*/ 0);

  // 4) x = gelu(ln(pre; per-subject params))
  ln_gelu_kernel<<<B_, 256, 0, stream>>>(pre, enc_ln_g, enc_ln_b, sid, nullptr,
                                         xbuf, H_);

  // 5) backbone: x = x + gelu(ln(x @ w_d + b_d))
  for (int d = 0; d < D_; ++d) {
    gemm_bf16<EPI_LINEAR><<<dim3(H_ / BN, gy), 256, 0, stream>>>(
        xbuf, bb_w + (size_t)d * H_ * H_, bb_b + (size_t)d * H_, nullptr, pre,
        B_, nullptr, H_, H_, H_, H_, 0, nullptr, 0, 0, 0);
    ln_gelu_kernel<<<B_, 256, 0, stream>>>(pre, bb_ln_g + (size_t)d * H_,
                                           bb_ln_b + (size_t)d * H_, nullptr,
                                           xbuf, xbuf, H_);
  }

  // 6) heads -> d_out = [img | txt]
  gemm_bf16<EPI_LINEAR><<<dim3(IMG_ / BN, gy), 256, 0, stream>>>(
      xbuf, img_w, img_b, nullptr, out, B_, nullptr, IMG_, H_, H_, IMG_, 0,
      nullptr, 0, 0, 0);
  gemm_bf16<EPI_LINEAR><<<dim3(TXT_ / BN, gy), 256, 0, stream>>>(
      xbuf, txt_w, txt_b, nullptr, out + (size_t)B_ * IMG_, B_, nullptr, TXT_,
      H_, H_, TXT_, 0, nullptr, 0, 0, 0);
}
